// VectorQuantizer_38920993636499
// MI455X (gfx1250) — matrix operand order repaired and checked by reading, not compiled, not run on hardware
//
#include <hip/hip_runtime.h>
#include <hip/hip_bf16.h>

// ---------------------------------------------------------------------------
// VQ-VAE vector quantizer for MI455X (gfx1250, wave32, WMMA).
//   N = 32768 rows, D = 256, K = 1024 codes.
//   S = X * E^T via v_wmma_f32_16x16x32_bf16 (f32 accumulate),
//   dist = ||x||^2 - 2 S + ||e||^2 (norm terms exact f32),
//   row argmin -> gather -> quantized + vq_loss.
// A-fragments live in registers (loaded straight from global with f32->bf16
// convert), embedding chunks double-buffered in LDS and streamed with
// GLOBAL_LOAD_ASYNC_TO_LDS_B128 (ASYNCcnt), nontemporal output stores.
// ---------------------------------------------------------------------------

typedef __attribute__((ext_vector_type(16))) __bf16 v16bf;
typedef __attribute__((ext_vector_type(8)))  float  v8f;
typedef __attribute__((ext_vector_type(4)))  float  v4f;
typedef int v4i __attribute__((vector_size(16)));

#define DIM        256
#define KCB        1024
#define ROWS_BLK   64
#define CHUNK      64
#define N_CHUNKS   (KCB / CHUNK)           // 16
#define N_ROWS     32768
#define Q_ELEMS    (N_ROWS * DIM)          // 8388608
#define N_BLOCKS   (N_ROWS / ROWS_BLK)     // 512

union Frag {
    v16bf  v;
    __bf16 e[16];
    int4   q[2];
};

// ---- async global->LDS copy (CDNA5 ASYNCcnt path) with sync fallback ------
#if __has_builtin(__builtin_amdgcn_global_load_async_to_lds_b128)
#define VQ_ASYNC_LDS 1
typedef __attribute__((address_space(1))) v4i GV4;
typedef __attribute__((address_space(3))) v4i LV4;
#endif

__device__ __forceinline__ void copy16_to_lds(const void* g, void* l) {
#ifdef VQ_ASYNC_LDS
    __builtin_amdgcn_global_load_async_to_lds_b128(
        (GV4*)(unsigned long long)g, (LV4*)l, 0, 0);
#else
    *(int4*)l = *(const int4*)g;
#endif
}

__device__ __forceinline__ void async_wait0() {
#ifdef VQ_ASYNC_LDS
#if __has_builtin(__builtin_amdgcn_s_wait_asynccnt)
    __builtin_amdgcn_s_wait_asynccnt(0);
#else
    asm volatile("s_wait_asynccnt 0x0" ::: "memory");
#endif
#endif
}

// ---------------------------------------------------------------------------
// Kernel 1: embedding f32 -> bf16 copy + per-row squared norms.
// ---------------------------------------------------------------------------
__global__ __launch_bounds__(256) void vq_prep(const float* __restrict__ emb,
                                               __bf16* __restrict__ ebf,
                                               float* __restrict__ enorm) {
    __shared__ float red[256];
    const int r = blockIdx.x;
    const int t = threadIdx.x;
    float v = emb[(size_t)r * DIM + t];
    ebf[(size_t)r * DIM + t] = (__bf16)v;
    red[t] = v * v;
    __syncthreads();
    #pragma unroll
    for (int s = 128; s > 0; s >>= 1) {
        if (t < s) red[t] += red[t + s];
        __syncthreads();
    }
    if (t == 0) enorm[r] = red[0];
}

// ---------------------------------------------------------------------------
// Kernel 2: main VQ kernel. One block = 64 input rows, 8 waves.
// Wave w: row-tile (w&3), column half (w>>2) of each 64-col chunk.
// ---------------------------------------------------------------------------
__global__ __launch_bounds__(256) void vq_main(const float* __restrict__ x,
                                               const float* __restrict__ emb,
                                               const __bf16* __restrict__ ebf,
                                               const float* __restrict__ enorm,
                                               float* __restrict__ out_q,
                                               float* __restrict__ out_idx,
                                               float* __restrict__ partials) {
    __shared__ __align__(16) __bf16 es[2][CHUNK][DIM];   // 64 KB double buffer
    __shared__ float xnorm[ROWS_BLK];
    __shared__ float candMin[2][ROWS_BLK];
    __shared__ int   candIdx[2][ROWS_BLK];
    __shared__ int   finalIdx[ROWS_BLK];
    __shared__ float red[256];

    const int t  = threadIdx.x;
    const int bm = blockIdx.x * ROWS_BLK;

    const int w       = t >> 5;     // wave 0..7
    const int L       = t & 31;     // lane
    const int h       = L >> 4;     // lane half
    const int n       = L & 15;     // M (A) / N (B) position
    const int rowTile = w & 3;      // 16-row tile of the 64
    const int colHalf = w >> 2;     // 32-col half of the chunk

    // ---- kick off async staging of embedding chunk 0 ----
    {
        const char* src = (const char*)ebf;           // chunk 0 base
        char* dst = (char*)&es[0][0][0];
        #pragma unroll
        for (int i = 0; i < 8; ++i) {
            const int off = (i * 256 + t) * 16;       // 2048 x 16B = 32 KB
            copy16_to_lds(src + off, dst + off);
        }
    }

    // ---- load all 8 A fragments for this wave's 16 rows into registers ----
    // A layout (16x32 bf16): lane L holds row M = L%16; lane-half h holds
    // K = kb+8h+{0..7} (VGPR0..3) and kb+16+8h+{0..7} (VGPR4..7).
    Frag afr[8];
    float sq = 0.f;
    {
        const float* xr = x + (size_t)(bm + rowTile * 16 + n) * DIM;
        #pragma unroll
        for (int s = 0; s < 8; ++s) {
            const int kb = 32 * s;
            v4f f0 = *(const v4f*)(xr + kb + 8 * h);
            v4f f1 = *(const v4f*)(xr + kb + 8 * h + 4);
            v4f f2 = *(const v4f*)(xr + kb + 16 + 8 * h);
            v4f f3 = *(const v4f*)(xr + kb + 16 + 8 * h + 4);
            afr[s].e[0]  = (__bf16)f0.x; afr[s].e[1]  = (__bf16)f0.y;
            afr[s].e[2]  = (__bf16)f0.z; afr[s].e[3]  = (__bf16)f0.w;
            afr[s].e[4]  = (__bf16)f1.x; afr[s].e[5]  = (__bf16)f1.y;
            afr[s].e[6]  = (__bf16)f1.z; afr[s].e[7]  = (__bf16)f1.w;
            afr[s].e[8]  = (__bf16)f2.x; afr[s].e[9]  = (__bf16)f2.y;
            afr[s].e[10] = (__bf16)f2.z; afr[s].e[11] = (__bf16)f2.w;
            afr[s].e[12] = (__bf16)f3.x; afr[s].e[13] = (__bf16)f3.y;
            afr[s].e[14] = (__bf16)f3.z; afr[s].e[15] = (__bf16)f3.w;
            v4f acc = f0 * f0 + f1 * f1 + f2 * f2 + f3 * f3;
            sq += acc.x + acc.y + acc.z + acc.w;
        }
        sq += __shfl_xor(sq, 16, 32);                 // combine the two halves
        if (w < 4 && h == 0) xnorm[rowTile * 16 + n] = sq;   // covers 64 rows
    }
    __syncthreads();

    // hoist row norms in C/D layout order (row = rowTile*16 + 8h + j)
    float xnr[8];
    #pragma unroll
    for (int j = 0; j < 8; ++j) xnr[j] = xnorm[rowTile * 16 + 8 * h + j];

    float runMin[2][8];
    int   runIdx[2][8];
    #pragma unroll
    for (int j = 0; j < 8; ++j) {
        runMin[0][j] = __builtin_inf();  runIdx[0][j] = 0;
        runMin[1][j] = __builtin_inf();  runIdx[1][j] = 0;
    }

    for (int chunk = 0; chunk < N_CHUNKS; ++chunk) {
        const int cur = chunk & 1;
        async_wait0();           // this wave's async loads for `cur` landed
        __syncthreads();         // everyone's landed; prev compute done

        if (chunk + 1 < N_CHUNKS) {   // prefetch next chunk into other buffer
            const char* src = (const char*)(ebf + (size_t)(chunk + 1) * CHUNK * DIM);
            char* dst = (char*)&es[cur ^ 1][0][0];
            #pragma unroll
            for (int i = 0; i < 8; ++i) {
                const int off = (i * 256 + t) * 16;
                copy16_to_lds(src + off, dst + off);
            }
        }

        // ---- S-tile accumulation over D=256 (2 wmma per 32-k step) ----
        const __bf16* er0 = &es[cur][colHalf * 32 + n][0];
        const __bf16* er1 = &es[cur][colHalf * 32 + 16 + n][0];
        v8f acc0 = {}, acc1 = {};
        #pragma unroll
        for (int s = 0; s < 8; ++s) {
            const int ks = 32 * s + 8 * h;    // B: K = 8h+{0..7}, 16+8h+{0..7}
            Frag b0, b1;
            b0.q[0] = *(const int4*)(er0 + ks);
            b0.q[1] = *(const int4*)(er0 + ks + 16);
            b1.q[0] = *(const int4*)(er1 + ks);
            b1.q[1] = *(const int4*)(er1 + ks + 16);
            acc0 = __builtin_amdgcn_wmma_f32_16x16x32_bf16(
                false, afr[s].v, false, b0.v, (short)0, acc0, false, false);
            acc1 = __builtin_amdgcn_wmma_f32_16x16x32_bf16(
                false, afr[s].v, false, b1.v, (short)0, acc1, false, false);
        }

        // ---- distances + running argmin update ----
        const int col0 = chunk * CHUNK + colHalf * 32 + n;
        const int col1 = col0 + 16;
        const float en0 = enorm[col0];
        const float en1 = enorm[col1];
        #pragma unroll
        for (int j = 0; j < 8; ++j) {
            const float d0 = xnr[j] - 2.f * acc0[j] + en0;
            const float d1 = xnr[j] - 2.f * acc1[j] + en1;
            if (d0 < runMin[0][j]) { runMin[0][j] = d0; runIdx[0][j] = col0; }
            if (d1 < runMin[1][j]) { runMin[1][j] = d1; runIdx[1][j] = col1; }
        }
    }

    // ---- merge the wave's two column tiles (same rows) ----
    float bv[8]; int bi[8];
    #pragma unroll
    for (int j = 0; j < 8; ++j) {
        bool take1 = (runMin[1][j] < runMin[0][j]) ||
                     (runMin[1][j] == runMin[0][j] && runIdx[1][j] < runIdx[0][j]);
        bv[j] = take1 ? runMin[1][j] : runMin[0][j];
        bi[j] = take1 ? runIdx[1][j] : runIdx[0][j];
    }
    // ---- butterfly argmin across the 16 lanes holding a row (wave32) ----
    #pragma unroll
    for (int mask = 1; mask < 16; mask <<= 1) {
        #pragma unroll
        for (int j = 0; j < 8; ++j) {
            float ov = __shfl_xor(bv[j], mask, 32);
            int   oi = __shfl_xor(bi[j], mask, 32);
            if (ov < bv[j] || (ov == bv[j] && oi < bi[j])) { bv[j] = ov; bi[j] = oi; }
        }
    }
    if (n == 0) {
        #pragma unroll
        for (int j = 0; j < 8; ++j) {
            const int r = rowTile * 16 + 8 * h + j;
            candMin[colHalf][r] = bv[j];
            candIdx[colHalf][r] = bi[j];
        }
    }
    __syncthreads();

    // ---- final per-row winner between the two column-half waves ----
    if (t < ROWS_BLK) {
        float v0 = candMin[0][t], v1 = candMin[1][t];
        int   i0 = candIdx[0][t], i1 = candIdx[1][t];
        int best = (v1 < v0 || (v1 == v0 && i1 < i0)) ? i1 : i0;
        finalIdx[t] = best;
        __builtin_nontemporal_store((float)best, out_idx + bm + t);
    }
    __syncthreads();

    // ---- gather embedding rows, straight-through output, loss partial ----
    float lsum = 0.f;
    {
        const int r  = t >> 2;
        const int cs = (t & 3) * 64;
        const int e  = finalIdx[r];
        const float* erow = emb + (size_t)e * DIM + cs;
        const float* xr   = x + (size_t)(bm + r) * DIM + cs;
        float* qr         = out_q + (size_t)(bm + r) * DIM + cs;
        #pragma unroll
        for (int c = 0; c < 64; c += 4) {
            v4f ev = *(const v4f*)(erow + c);
            v4f xv = *(const v4f*)(xr + c);
            // straight-through forward value: x + (q - x), matching JAX rounding
            v4f d  = ev - xv;
            v4f qv = xv + d;
            __builtin_nontemporal_store(qv, (v4f*)(qr + c));
            v4f e2 = (qv - xv) * (qv - xv);
            lsum += e2.x + e2.y + e2.z + e2.w;
        }
    }
    red[t] = lsum;
    __syncthreads();
    #pragma unroll
    for (int s = 128; s > 0; s >>= 1) {
        if (t < s) red[t] += red[t + s];
        __syncthreads();
    }
    if (t == 0) partials[blockIdx.x] = red[0];
}

// ---------------------------------------------------------------------------
// Kernel 3: deterministic reduction of 512 block partials -> vq_loss scalar.
// vq_loss = 1.25 * sum((q-x)^2) / (N*D)
// ---------------------------------------------------------------------------
__global__ __launch_bounds__(512) void vq_loss_reduce(const float* __restrict__ partials,
                                                      float* __restrict__ out_loss) {
    __shared__ float buf[512];
    const int t = threadIdx.x;
    buf[t] = partials[t];
    __syncthreads();
    #pragma unroll
    for (int s = 256; s > 0; s >>= 1) {
        if (t < s) buf[t] += buf[t + s];
        __syncthreads();
    }
    if (t == 0) *out_loss = buf[0] * (1.25f / (float)Q_ELEMS);
}

// ---------------------------------------------------------------------------
extern "C" void kernel_launch(void* const* d_in, const int* in_sizes, int n_in,
                              void* d_out, int out_size, void* d_ws, size_t ws_size,
                              hipStream_t stream) {
    const float* x   = (const float*)d_in[0];   // (32,32,32,256) f32
    const float* emb = (const float*)d_in[1];   // (1024,256) f32

    // workspace layout
    __bf16* ebf     = (__bf16*)d_ws;                                  // 512 KB
    float*  enorm   = (float*)((char*)d_ws + (size_t)KCB * DIM * 2);  // 4 KB
    float*  partial = enorm + KCB;                                    // 2 KB

    float* out = (float*)d_out;
    float* out_q    = out;                 // 8388608 f32
    float* out_loss = out + Q_ELEMS;       // 1 f32
    float* out_idx  = out + Q_ELEMS + 1;   // 32768 f32 (index values)

    vq_prep<<<KCB, 256, 0, stream>>>(emb, ebf, enorm);
    vq_main<<<N_BLOCKS, 256, 0, stream>>>(x, emb, ebf, enorm,
                                          out_q, out_idx, partial);
    vq_loss_reduce<<<1, 512, 0, stream>>>(partial, out_loss);
}
